// Seq2Seq_33732673143693
// MI455X (gfx1250) — compile-verified, matrix operand order checked by reading, not run on hardware
//
#include <hip/hip_runtime.h>
#include <hip/hip_bf16.h>
#include <math.h>

// Problem constants (from reference)
#define S_ 64
#define T_ 48
#define B_ 32
#define VS_ 32000
#define VT_ 32000
#define E_ 256
#define H_ 512
#define D_ 512

typedef __attribute__((ext_vector_type(16))) __bf16 v16bf;
typedef __attribute__((ext_vector_type(8)))  __bf16 v8bf;
typedef __attribute__((ext_vector_type(8)))  float  v8f;
typedef __attribute__((ext_vector_type(4)))  float  f4v;

__device__ __forceinline__ float sigf(float x) { return 1.0f / (1.0f + __expf(-x)); }

// ---------------------------------------------------------------------------
// Main GEMM: C[M,N] = act(A[M,K] @ Wb.T + bias[N]), Wb is bf16 [N,K] row-major.
// One wave computes a 16(M) x 64(N) strip. Per K-step: 4 independent float4
// loads for A (converted in-register) + 8 independent b128 bf16 loads for B
// (no conversion), then 4 back-to-back v_wmma_f32_16x16x32_bf16.
// Fragment layouts per CDNA5 ISA 7.12.2 (wave32). All N used are %64 == 0,
// all K are %32 == 0.
// ---------------------------------------------------------------------------
template <int ACT>
__global__ void gemm_bt_bf16(const float* __restrict__ A,
                             const __bf16* __restrict__ Bm,
                             const float* __restrict__ bias,
                             float* __restrict__ C,
                             int M, int N, int K) {
  const int lane = threadIdx.x & 31;
  const int hi   = lane >> 4;   // 0 -> K {0..7,16..23}, 1 -> K {8..15,24..31}
  const int r    = lane & 15;
  const int n0   = blockIdx.x * 64;
  const int m0   = blockIdx.y * 16;

  v8f acc[4];
#pragma unroll
  for (int q = 0; q < 4; ++q) {
    const float bval = bias ? bias[n0 + 16 * q + r] : 0.0f;
#pragma unroll
    for (int i = 0; i < 8; ++i) acc[q][i] = bval;
  }

  const float*  Ap = A + (size_t)(m0 + r) * K + 8 * hi;
  const __bf16* Bp0 = Bm + (size_t)(n0 +  0 + r) * K + 8 * hi;
  const __bf16* Bp1 = Bm + (size_t)(n0 + 16 + r) * K + 8 * hi;
  const __bf16* Bp2 = Bm + (size_t)(n0 + 32 + r) * K + 8 * hi;
  const __bf16* Bp3 = Bm + (size_t)(n0 + 48 + r) * K + 8 * hi;

  for (int k0 = 0; k0 < K; k0 += 32) {
    if (k0 + 32 < K) {
      __builtin_prefetch(Ap + k0 + 32, 0, 0);     // global_prefetch
      __builtin_prefetch(Bp0 + k0 + 32, 0, 0);
    }
    // Issue all loads first (independent destinations), then convert/compute.
    f4v a0 = *(const f4v*)(Ap + k0);
    f4v a1 = *(const f4v*)(Ap + k0 + 4);
    f4v a2 = *(const f4v*)(Ap + k0 + 16);
    f4v a3 = *(const f4v*)(Ap + k0 + 20);
    v8bf b0lo = *(const v8bf*)(Bp0 + k0), b0hi = *(const v8bf*)(Bp0 + k0 + 16);
    v8bf b1lo = *(const v8bf*)(Bp1 + k0), b1hi = *(const v8bf*)(Bp1 + k0 + 16);
    v8bf b2lo = *(const v8bf*)(Bp2 + k0), b2hi = *(const v8bf*)(Bp2 + k0 + 16);
    v8bf b3lo = *(const v8bf*)(Bp3 + k0), b3hi = *(const v8bf*)(Bp3 + k0 + 16);

    v16bf af;
#pragma unroll
    for (int j = 0; j < 4; ++j) {
      af[j]      = (__bf16)a0[j];
      af[4 + j]  = (__bf16)a1[j];
      af[8 + j]  = (__bf16)a2[j];
      af[12 + j] = (__bf16)a3[j];
    }
    v16bf bf0, bf1, bf2, bf3;
#pragma unroll
    for (int j = 0; j < 8; ++j) {
      bf0[j] = b0lo[j]; bf0[8 + j] = b0hi[j];
      bf1[j] = b1lo[j]; bf1[8 + j] = b1hi[j];
      bf2[j] = b2lo[j]; bf2[8 + j] = b2hi[j];
      bf3[j] = b3lo[j]; bf3[8 + j] = b3hi[j];
    }
    acc[0] = __builtin_amdgcn_wmma_f32_16x16x32_bf16(false, af, false, bf0, (short)0, acc[0], false, false);
    acc[1] = __builtin_amdgcn_wmma_f32_16x16x32_bf16(false, af, false, bf1, (short)0, acc[1], false, false);
    acc[2] = __builtin_amdgcn_wmma_f32_16x16x32_bf16(false, af, false, bf2, (short)0, acc[2], false, false);
    acc[3] = __builtin_amdgcn_wmma_f32_16x16x32_bf16(false, af, false, bf3, (short)0, acc[3], false, false);
  }

#pragma unroll
  for (int q = 0; q < 4; ++q) {
#pragma unroll
    for (int i = 0; i < 8; ++i) {
      float v = acc[q][i];
      if (ACT == 1) v = tanhf(v);
      C[(size_t)(m0 + i + 8 * hi) * N + n0 + 16 * q + r] = v;  // C: VGPR i -> M = i + 8*hi
    }
  }
}

// ---------------------------------------------------------------------------
// Fallback GEMM (float weights, used only if ws too small for bf16 copies):
//   BT=true : B is W[N,K] row-major;  BT=false: B is W[K,N] row-major.
// ---------------------------------------------------------------------------
template <bool BT, int ACT>
__global__ void gemm_wmma_f32b(const float* __restrict__ A,
                               const float* __restrict__ Bm,
                               const float* __restrict__ bias,
                               float* __restrict__ C,
                               int M, int N, int K) {
  const int lane = threadIdx.x & 31;
  const int hi   = lane >> 4;
  const int r    = lane & 15;
  const int n0   = blockIdx.x * 64;
  const int m0   = blockIdx.y * 16;

  v8f acc[4];
#pragma unroll
  for (int q = 0; q < 4; ++q) {
    const float bval = bias ? bias[n0 + 16 * q + r] : 0.0f;
#pragma unroll
    for (int i = 0; i < 8; ++i) acc[q][i] = bval;
  }
  const float* Arow = A + (size_t)(m0 + r) * K;

  for (int k0 = 0; k0 < K; k0 += 32) {
    v16bf af;
#pragma unroll
    for (int j = 0; j < 16; ++j) {
      const int k = k0 + j + ((j < 8) ? 0 : 8) + 8 * hi;
      af[j] = (__bf16)Arow[k];
    }
#pragma unroll
    for (int q = 0; q < 4; ++q) {
      v16bf bfv;
#pragma unroll
      for (int j = 0; j < 16; ++j) {
        const int k = k0 + j + ((j < 8) ? 0 : 8) + 8 * hi;
        bfv[j] = BT ? (__bf16)Bm[(size_t)(n0 + 16 * q + r) * K + k]
                    : (__bf16)Bm[(size_t)k * N + n0 + 16 * q + r];
      }
      acc[q] = __builtin_amdgcn_wmma_f32_16x16x32_bf16(false, af, false, bfv,
                                                       (short)0, acc[q], false, false);
    }
  }
#pragma unroll
  for (int q = 0; q < 4; ++q) {
#pragma unroll
    for (int i = 0; i < 8; ++i) {
      float v = acc[q][i];
      if (ACT == 1) v = tanhf(v);
      C[(size_t)(m0 + i + 8 * hi) * N + n0 + 16 * q + r] = v;
    }
  }
}

// ---------------------------------------------------------------------------
// Weight preparation
// ---------------------------------------------------------------------------
// Straight f32 -> bf16 (weight already [N,K])
__global__ void f32_to_bf16_k(const float* __restrict__ in, __bf16* __restrict__ out, size_t n) {
  size_t i = (size_t)blockIdx.x * 256 + threadIdx.x;
  if (i < n) out[i] = (__bf16)in[i];
}

// Transpose-convert: in [K,N] f32 -> out [N,K] bf16, via LDS tile (coalesced
// loads and stores). K%32==0, N%32==0.
__global__ void transpose_bf16_k(const float* __restrict__ in, __bf16* __restrict__ out,
                                 int K, int N) {
  __shared__ __bf16 tile[32][33];
  const int nb = blockIdx.x * 32;
  const int kb = blockIdx.y * 32;
  for (int yy = threadIdx.y; yy < 32; yy += 8)
    tile[yy][threadIdx.x] = (__bf16)in[(size_t)(kb + yy) * N + nb + threadIdx.x];
  __syncthreads();
  for (int yy = threadIdx.y; yy < 32; yy += 8)
    out[(size_t)(nb + yy) * K + kb + threadIdx.x] = tile[threadIdx.x][yy];
}

// ---------------------------------------------------------------------------
// Small helper kernels
// ---------------------------------------------------------------------------
__global__ void zero_f32_k(float* p, size_t n) {
  size_t i = (size_t)blockIdx.x * 256 + threadIdx.x;
  if (i < n) p[i] = 0.0f;
}

__global__ void gather_rows_k(const float* __restrict__ emb, const int* __restrict__ idx,
                              float* __restrict__ out, int ncols) {
  const int row = blockIdx.x;
  const int src = idx[row];
  for (int c = threadIdx.x; c < ncols; c += blockDim.x)
    out[(size_t)row * ncols + c] = emb[(size_t)src * ncols + c];
}

// h_new = (1-z)*n + z*h (in-place on h); optional scatter into enc_bs layout
__global__ void gru_pointwise_k(const float* __restrict__ gi, const float* __restrict__ gh,
                                float* __restrict__ h, float* __restrict__ extra,
                                int extra_stride, int Hh) {
  int i = blockIdx.x * blockDim.x + threadIdx.x;
  if (i >= B_ * Hh) return;
  int b = i / Hh, d = i - b * Hh;
  const float* gib = gi + (size_t)b * 3 * Hh;
  const float* ghb = gh + (size_t)b * 3 * Hh;
  float rr = sigf(gib[d] + ghb[d]);
  float zz = sigf(gib[Hh + d] + ghb[Hh + d]);
  float nn = tanhf(gib[2 * Hh + d] + rr * ghb[2 * Hh + d]);
  float out = (1.0f - zz) * nn + zz * h[i];
  h[i] = out;
  if (extra) extra[(size_t)b * extra_stride + d] = out;
}

__global__ void concat2_k(const float* __restrict__ a, int wa,
                          const float* __restrict__ b, int wb,
                          float* __restrict__ out, int rows) {
  int w = wa + wb;
  int i = blockIdx.x * blockDim.x + threadIdx.x;
  if (i >= rows * w) return;
  int rr = i / w, c = i - rr * w;
  out[i] = (c < wa) ? a[(size_t)rr * wa + c] : b[(size_t)rr * wb + (c - wa)];
}

__global__ void concat3_k(const float* __restrict__ a, int wa,
                          const float* __restrict__ b, int wb,
                          const float* __restrict__ c, int wc,
                          float* __restrict__ out, int rows) {
  int w = wa + wb + wc;
  int i = blockIdx.x * blockDim.x + threadIdx.x;
  if (i >= rows * w) return;
  int rr = i / w, cc = i - rr * w;
  float v;
  if (cc < wa)            v = a[(size_t)rr * wa + cc];
  else if (cc < wa + wb)  v = b[(size_t)rr * wb + (cc - wa)];
  else                    v = c[(size_t)rr * wc + (cc - wa - wb)];
  out[i] = v;
}

// cat[b,s,:] = concat(h[b,:] (D), enc_bs[b,s,:] (2H))
__global__ void build_attn_cat_k(const float* __restrict__ h, const float* __restrict__ enc_bs,
                                 float* __restrict__ out) {
  const int W = D_ + 2 * H_;  // 1536
  int i = blockIdx.x * blockDim.x + threadIdx.x;
  if (i >= B_ * S_ * W) return;
  int bs = i / W, c = i - bs * W;
  int b = bs / S_, s = bs - b * S_;
  out[i] = (c < D_) ? h[(size_t)b * D_ + c]
                    : enc_bs[((size_t)b * S_ + s) * (2 * H_) + (c - D_)];
}

// scores[b,s] = energy[b,s,:] . v
__global__ void attn_scores_k(const float* __restrict__ energy, const float* __restrict__ v,
                              float* __restrict__ scores) {
  int i = blockIdx.x * blockDim.x + threadIdx.x;
  if (i >= B_ * S_) return;
  const float* e = energy + (size_t)i * D_;
  float sum = 0.0f;
  for (int d = 0; d < D_; ++d) sum += e[d] * v[d];
  scores[i] = sum;
}

__global__ void softmax_rows_k(float* __restrict__ scores) {
  int b = threadIdx.x;
  if (b >= B_) return;
  float* s = scores + (size_t)b * S_;
  float m = s[0];
  for (int i = 1; i < S_; ++i) m = fmaxf(m, s[i]);
  float sum = 0.0f;
  for (int i = 0; i < S_; ++i) { float e = __expf(s[i] - m); s[i] = e; sum += e; }
  float inv = 1.0f / sum;
  for (int i = 0; i < S_; ++i) s[i] *= inv;
}

// weighted[b,h] = sum_s a[b,s] * enc_bs[b,s,h]
__global__ void attn_weighted_k(const float* __restrict__ a, const float* __restrict__ enc_bs,
                                float* __restrict__ weighted) {
  int i = blockIdx.x * blockDim.x + threadIdx.x;
  if (i >= B_ * 2 * H_) return;
  int b = i / (2 * H_), hcol = i - b * 2 * H_;
  const float* ab = a + (size_t)b * S_;
  const float* eb = enc_bs + (size_t)b * S_ * 2 * H_ + hcol;
  float sum = 0.0f;
  for (int s = 0; s < S_; ++s) sum += ab[s] * eb[(size_t)s * 2 * H_];
  weighted[i] = sum;
}

// next_inp[b] = tf_mask[t] > 0 ? trg[t,b] : argmax_v pred[b,v]  (first max wins)
__global__ void argmax_next_k(const float* __restrict__ pred, const int* __restrict__ trg,
                              const int* __restrict__ tf_mask, int t, int* __restrict__ next_inp) {
  __shared__ float sv[256];
  __shared__ int   si[256];
  const int b = blockIdx.x;
  const float* p = pred + (size_t)b * VT_;
  float best = -INFINITY;
  int   bi   = 0x7fffffff;
  for (int i = threadIdx.x; i < VT_; i += 256) {
    float v = p[i];
    if (v > best) { best = v; bi = i; }
  }
  sv[threadIdx.x] = best; si[threadIdx.x] = bi;
  __syncthreads();
  for (int off = 128; off > 0; off >>= 1) {
    if ((int)threadIdx.x < off) {
      float v2 = sv[threadIdx.x + off]; int i2 = si[threadIdx.x + off];
      if (v2 > sv[threadIdx.x] || (v2 == sv[threadIdx.x] && i2 < si[threadIdx.x])) {
        sv[threadIdx.x] = v2; si[threadIdx.x] = i2;
      }
    }
    __syncthreads();
  }
  if (threadIdx.x == 0)
    next_inp[b] = (tf_mask[t] > 0) ? trg[(size_t)t * B_ + b] : si[0];
}

__global__ void init_inp_k(const int* __restrict__ trg, int* __restrict__ next_inp) {
  int b = threadIdx.x;
  if (b < B_) next_inp[b] = trg[b];
}

// ---------------------------------------------------------------------------
// Host orchestration
// ---------------------------------------------------------------------------
static inline dim3 gemm_grid(int M, int N) { return dim3(N / 64, M / 16); }

extern "C" void kernel_launch(void* const* d_in, const int* in_sizes, int n_in,
                              void* d_out, int out_size, void* d_ws, size_t ws_size,
                              hipStream_t stream) {
  const int*   src       = (const int*)d_in[0];
  const int*   trg       = (const int*)d_in[1];
  const int*   tf_mask   = (const int*)d_in[2];
  const float* enc_emb   = (const float*)d_in[3];
  const float* Wih_f     = (const float*)d_in[4];
  const float* Whh_f     = (const float*)d_in[5];
  const float* bih_f     = (const float*)d_in[6];
  const float* bhh_f     = (const float*)d_in[7];
  const float* Wih_b     = (const float*)d_in[8];
  const float* Whh_b     = (const float*)d_in[9];
  const float* bih_b     = (const float*)d_in[10];
  const float* bhh_b     = (const float*)d_in[11];
  const float* Wfc       = (const float*)d_in[12];
  const float* bfc       = (const float*)d_in[13];
  const float* attn_W    = (const float*)d_in[14];
  const float* attn_b    = (const float*)d_in[15];
  const float* attn_v    = (const float*)d_in[16];
  const float* dec_emb   = (const float*)d_in[17];
  const float* dec_Wih   = (const float*)d_in[18];
  const float* dec_Whh   = (const float*)d_in[19];
  const float* dec_bih   = (const float*)d_in[20];
  const float* dec_bhh   = (const float*)d_in[21];
  const float* out_W     = (const float*)d_in[22];
  const float* out_b     = (const float*)d_in[23];
  float* out = (float*)d_out;

  const int KO = 2 * H_ + D_ + E_;  // 1792

  // --- workspace layout ---
  char* w = (char*)d_ws;
  size_t used = 0;
  auto alloc = [&](size_t bytes) -> void* {
    void* p = w + used;
    used += (bytes + 255) & ~(size_t)255;
    return p;
  };
  float* emb_src  = (float*)alloc((size_t)S_ * B_ * E_ * 4);
  float* gi_f     = (float*)alloc((size_t)S_ * B_ * 3 * H_ * 4);
  float* gi_b     = (float*)alloc((size_t)S_ * B_ * 3 * H_ * 4);
  float* h_f      = (float*)alloc((size_t)B_ * H_ * 4);
  float* h_b      = (float*)alloc((size_t)B_ * H_ * 4);
  float* gh_f     = (float*)alloc((size_t)B_ * 3 * H_ * 4);
  float* gh_b     = (float*)alloc((size_t)B_ * 3 * H_ * 4);
  float* enc_bs   = (float*)alloc((size_t)B_ * S_ * 2 * H_ * 4);
  float* hcat     = (float*)alloc((size_t)B_ * 2 * H_ * 4);
  float* h_dec    = (float*)alloc((size_t)B_ * D_ * 4);
  float* cat_attn = (float*)alloc((size_t)B_ * S_ * (D_ + 2 * H_) * 4);
  float* energy   = (float*)alloc((size_t)B_ * S_ * D_ * 4);
  float* scores   = (float*)alloc((size_t)B_ * S_ * 4);
  float* weighted = (float*)alloc((size_t)B_ * 2 * H_ * 4);
  float* emb_t    = (float*)alloc((size_t)B_ * E_ * 4);
  float* rnn_in   = (float*)alloc((size_t)B_ * (E_ + 2 * H_) * 4);
  float* gi_dec   = (float*)alloc((size_t)B_ * 3 * D_ * 4);
  float* gh_dec   = (float*)alloc((size_t)B_ * 3 * D_ * 4);
  float* pred_cat = (float*)alloc((size_t)B_ * KO * 4);
  int*   next_inp = (int*)alloc((size_t)B_ * 4);
  // bf16 weight copies, all in [N,K] layout for the BT GEMM
  __bf16* bWih_f  = (__bf16*)alloc((size_t)3 * H_ * E_ * 2);
  __bf16* bWih_b  = (__bf16*)alloc((size_t)3 * H_ * E_ * 2);
  __bf16* bWhh_f  = (__bf16*)alloc((size_t)3 * H_ * H_ * 2);
  __bf16* bWhh_b  = (__bf16*)alloc((size_t)3 * H_ * H_ * 2);
  __bf16* bWfcT   = (__bf16*)alloc((size_t)D_ * 2 * H_ * 2);          // [D, 2H]
  __bf16* battnWT = (__bf16*)alloc((size_t)D_ * (D_ + 2 * H_) * 2);   // [D, 1536]
  __bf16* bdecWih = (__bf16*)alloc((size_t)3 * D_ * (E_ + 2 * H_) * 2);
  __bf16* bdecWhh = (__bf16*)alloc((size_t)3 * D_ * D_ * 2);
  __bf16* boutWT  = (__bf16*)alloc((size_t)VT_ * KO * 2);             // [32000, 1792]
  const bool use_bf16 = (used <= ws_size);  // deterministic fallback

  const dim3 tb(32, 8);
  if (use_bf16) {
    // straight conversions (already [N,K])
    f32_to_bf16_k<<<dim3((3 * H_ * E_ + 255) / 256), 256, 0, stream>>>(Wih_f, bWih_f, (size_t)3 * H_ * E_);
    f32_to_bf16_k<<<dim3((3 * H_ * E_ + 255) / 256), 256, 0, stream>>>(Wih_b, bWih_b, (size_t)3 * H_ * E_);
    f32_to_bf16_k<<<dim3((3 * H_ * H_ + 255) / 256), 256, 0, stream>>>(Whh_f, bWhh_f, (size_t)3 * H_ * H_);
    f32_to_bf16_k<<<dim3((3 * H_ * H_ + 255) / 256), 256, 0, stream>>>(Whh_b, bWhh_b, (size_t)3 * H_ * H_);
    f32_to_bf16_k<<<dim3((3 * D_ * (E_ + 2 * H_) + 255) / 256), 256, 0, stream>>>(dec_Wih, bdecWih, (size_t)3 * D_ * (E_ + 2 * H_));
    f32_to_bf16_k<<<dim3((3 * D_ * D_ + 255) / 256), 256, 0, stream>>>(dec_Whh, bdecWhh, (size_t)3 * D_ * D_);
    // transpose-convert ([K,N] -> [N,K])
    transpose_bf16_k<<<dim3(D_ / 32, 2 * H_ / 32), tb, 0, stream>>>(Wfc, bWfcT, 2 * H_, D_);
    transpose_bf16_k<<<dim3(D_ / 32, (D_ + 2 * H_) / 32), tb, 0, stream>>>(attn_W, battnWT, D_ + 2 * H_, D_);
    transpose_bf16_k<<<dim3(VT_ / 32, KO / 32), tb, 0, stream>>>(out_W, boutWT, KO, VT_);
  }

  // --- encoder ---
  gather_rows_k<<<dim3(S_ * B_), 256, 0, stream>>>(enc_emb, src, emb_src, E_);
  if (use_bf16) {
    gemm_bt_bf16<0><<<gemm_grid(S_ * B_, 3 * H_), 32, 0, stream>>>(emb_src, bWih_f, bih_f, gi_f, S_ * B_, 3 * H_, E_);
    gemm_bt_bf16<0><<<gemm_grid(S_ * B_, 3 * H_), 32, 0, stream>>>(emb_src, bWih_b, bih_b, gi_b, S_ * B_, 3 * H_, E_);
  } else {
    gemm_wmma_f32b<true, 0><<<gemm_grid(S_ * B_, 3 * H_), 32, 0, stream>>>(emb_src, Wih_f, bih_f, gi_f, S_ * B_, 3 * H_, E_);
    gemm_wmma_f32b<true, 0><<<gemm_grid(S_ * B_, 3 * H_), 32, 0, stream>>>(emb_src, Wih_b, bih_b, gi_b, S_ * B_, 3 * H_, E_);
  }
  zero_f32_k<<<dim3((B_ * H_ + 255) / 256), 256, 0, stream>>>(h_f, (size_t)B_ * H_);
  zero_f32_k<<<dim3((B_ * H_ + 255) / 256), 256, 0, stream>>>(h_b, (size_t)B_ * H_);

  const int es = S_ * 2 * H_;
  for (int t = 0; t < S_; ++t) {
    if (use_bf16) {
      gemm_bt_bf16<0><<<gemm_grid(B_, 3 * H_), 32, 0, stream>>>(h_f, bWhh_f, bhh_f, gh_f, B_, 3 * H_, H_);
    } else {
      gemm_wmma_f32b<true, 0><<<gemm_grid(B_, 3 * H_), 32, 0, stream>>>(h_f, Whh_f, bhh_f, gh_f, B_, 3 * H_, H_);
    }
    gru_pointwise_k<<<dim3((B_ * H_ + 255) / 256), 256, 0, stream>>>(
        gi_f + (size_t)t * B_ * 3 * H_, gh_f, h_f, enc_bs + (size_t)t * 2 * H_, es, H_);
    if (use_bf16) {
      gemm_bt_bf16<0><<<gemm_grid(B_, 3 * H_), 32, 0, stream>>>(h_b, bWhh_b, bhh_b, gh_b, B_, 3 * H_, H_);
    } else {
      gemm_wmma_f32b<true, 0><<<gemm_grid(B_, 3 * H_), 32, 0, stream>>>(h_b, Whh_b, bhh_b, gh_b, B_, 3 * H_, H_);
    }
    gru_pointwise_k<<<dim3((B_ * H_ + 255) / 256), 256, 0, stream>>>(
        gi_b + (size_t)(S_ - 1 - t) * B_ * 3 * H_, gh_b, h_b,
        enc_bs + (size_t)(S_ - 1 - t) * 2 * H_ + H_, es, H_);
  }
  concat2_k<<<dim3((B_ * 2 * H_ + 255) / 256), 256, 0, stream>>>(h_f, H_, h_b, H_, hcat, B_);
  if (use_bf16) {
    gemm_bt_bf16<1><<<gemm_grid(B_, D_), 32, 0, stream>>>(hcat, bWfcT, bfc, h_dec, B_, D_, 2 * H_);
  } else {
    gemm_wmma_f32b<false, 1><<<gemm_grid(B_, D_), 32, 0, stream>>>(hcat, Wfc, bfc, h_dec, B_, D_, 2 * H_);
  }

  // --- decoder ---
  zero_f32_k<<<dim3((unsigned)(((size_t)B_ * VT_ + 255) / 256)), 256, 0, stream>>>(out, (size_t)B_ * VT_);
  init_inp_k<<<dim3(1), 32, 0, stream>>>(trg, next_inp);

  for (int t = 1; t < T_; ++t) {
    float* pred = out + (size_t)t * B_ * VT_;
    gather_rows_k<<<dim3(B_), 256, 0, stream>>>(dec_emb, next_inp, emb_t, E_);
    build_attn_cat_k<<<dim3((B_ * S_ * (D_ + 2 * H_) + 255) / 256), 256, 0, stream>>>(h_dec, enc_bs, cat_attn);
    if (use_bf16) {
      gemm_bt_bf16<1><<<gemm_grid(B_ * S_, D_), 32, 0, stream>>>(cat_attn, battnWT, attn_b, energy, B_ * S_, D_, D_ + 2 * H_);
    } else {
      gemm_wmma_f32b<false, 1><<<gemm_grid(B_ * S_, D_), 32, 0, stream>>>(cat_attn, attn_W, attn_b, energy, B_ * S_, D_, D_ + 2 * H_);
    }
    attn_scores_k<<<dim3((B_ * S_ + 255) / 256), 256, 0, stream>>>(energy, attn_v, scores);
    softmax_rows_k<<<dim3(1), 32, 0, stream>>>(scores);
    attn_weighted_k<<<dim3((B_ * 2 * H_ + 255) / 256), 256, 0, stream>>>(scores, enc_bs, weighted);
    concat2_k<<<dim3((B_ * (E_ + 2 * H_) + 255) / 256), 256, 0, stream>>>(emb_t, E_, weighted, 2 * H_, rnn_in, B_);
    if (use_bf16) {
      gemm_bt_bf16<0><<<gemm_grid(B_, 3 * D_), 32, 0, stream>>>(rnn_in, bdecWih, dec_bih, gi_dec, B_, 3 * D_, E_ + 2 * H_);
      gemm_bt_bf16<0><<<gemm_grid(B_, 3 * D_), 32, 0, stream>>>(h_dec, bdecWhh, dec_bhh, gh_dec, B_, 3 * D_, D_);
    } else {
      gemm_wmma_f32b<true, 0><<<gemm_grid(B_, 3 * D_), 32, 0, stream>>>(rnn_in, dec_Wih, dec_bih, gi_dec, B_, 3 * D_, E_ + 2 * H_);
      gemm_wmma_f32b<true, 0><<<gemm_grid(B_, 3 * D_), 32, 0, stream>>>(h_dec, dec_Whh, dec_bhh, gh_dec, B_, 3 * D_, D_);
    }
    gru_pointwise_k<<<dim3((B_ * D_ + 255) / 256), 256, 0, stream>>>(gi_dec, gh_dec, h_dec, (float*)nullptr, 0, D_);
    concat3_k<<<dim3((B_ * KO + 255) / 256), 256, 0, stream>>>(h_dec, D_, weighted, 2 * H_, emb_t, E_, pred_cat, B_);
    if (use_bf16) {
      gemm_bt_bf16<0><<<gemm_grid(B_, VT_), 32, 0, stream>>>(pred_cat, boutWT, out_b, pred, B_, VT_, KO);
    } else {
      gemm_wmma_f32b<false, 0><<<gemm_grid(B_, VT_), 32, 0, stream>>>(pred_cat, out_W, out_b, pred, B_, VT_, KO);
    }
    argmax_next_k<<<dim3(B_), 256, 0, stream>>>(pred, trg, tf_mask, t, next_inp);
  }
}
